// GraphTransformerEmbedding_20899310862729
// MI455X (gfx1250) — matrix operand order repaired and checked by reading, not compiled, not run on hardware
//
#include <hip/hip_runtime.h>
#include <hip/hip_bf16.h>

typedef __attribute__((ext_vector_type(16))) _Float16 v16h;
typedef __attribute__((ext_vector_type(8)))  _Float16 v8h;
typedef __attribute__((ext_vector_type(8)))  float    v8f;
typedef __attribute__((ext_vector_type(4)))  unsigned int u32x4;

#define NHID 32
#define NHEADS 8
#define NBT 16384
#define NROWS (NBT*17)            // 278528
#define NMTILES (NROWS/16)        // 17408
#define WPB 8                     // waves per block
#define MT_BLOCKS (NMTILES/WPB)   // 2176

// adjacency bitmasks for the 17-node graph (bit j of ADJ_MASK[i] = A[i][j])
__constant__ unsigned ADJ_MASK[17] = {
    0x6u, 0x9u, 0x11u, 0x22u, 0x44u, 0x8C8u, 0x1130u, 0x220u, 0x440u,
    0x80u, 0x100u, 0x3020u, 0x4840u, 0x8800u, 0x11000u, 0x2000u, 0x4000u
};

// ---------------------------------------------------------------- WMMA helpers

__device__ __forceinline__ v8f wmma16(v16h a, v16h b, v8f c) {
    // v_wmma_f32_16x16x32_f16 : D = A(16x32 f16) * B(32x16 f16) + C(f32)
    return __builtin_amdgcn_wmma_f32_16x16x32_f16(false, a, false, b, (short)0, c, false, false);
}

__device__ __forceinline__ v8f zero8() {
    v8f c = {0.f, 0.f, 0.f, 0.f, 0.f, 0.f, 0.f, 0.f};
    return c;
}

// ------------------------------------------------- async global->LDS staging
// GLOBAL_LOAD_ASYNC_TO_LDS_B128 (ASYNCcnt-tracked); each lane moves 16 bytes.
// All call sites use uniform trip counts (bytes multiple of 4096 per block).
__device__ __forceinline__ void stage_to_lds(_Float16* sdst, const _Float16* gsrc,
                                             int bytes, int tid) {
    uint32_t lbase = (uint32_t)(uintptr_t)sdst;          // low 32 bits = LDS offset
    unsigned long long gbase = (unsigned long long)(uintptr_t)gsrc;
    int chunks = bytes >> 4;
    for (int idx = tid; idx < chunks; idx += 256) {
        asm volatile("global_load_async_to_lds_b128 %0, %1, off"
                     :: "v"(lbase + (uint32_t)(idx << 4)),
                        "v"(gbase + (unsigned long long)(idx << 4))
                     : "memory");
    }
#if __has_builtin(__builtin_amdgcn_s_wait_asynccnt)
    __builtin_amdgcn_s_wait_asynccnt(0);
#else
    asm volatile("s_wait_asynccnt 0x0" ::: "memory");
#endif
    __syncthreads();
}

// ------------------------------------------------- fragment loads

// A fragment (16x32, 16-bit layout): lane l (row=l&15) holds K = kb+{0..7}, kb+16+{0..7}, kb=8*(l>>4)
__device__ __forceinline__ v16h load_a_frag_f32(const float* base, int ld, int lane) {
    int row = lane & 15, kb = (lane >> 4) * 8;
    const float* p = base + (size_t)row * ld + kb;
    float4 c0 = *(const float4*)(p + 0);
    float4 c1 = *(const float4*)(p + 4);
    float4 c2 = *(const float4*)(p + 16);
    float4 c3 = *(const float4*)(p + 20);
    v16h a;
    a[0]=(_Float16)c0.x; a[1]=(_Float16)c0.y; a[2]=(_Float16)c0.z; a[3]=(_Float16)c0.w;
    a[4]=(_Float16)c1.x; a[5]=(_Float16)c1.y; a[6]=(_Float16)c1.z; a[7]=(_Float16)c1.w;
    a[8]=(_Float16)c2.x; a[9]=(_Float16)c2.y; a[10]=(_Float16)c2.z; a[11]=(_Float16)c2.w;
    a[12]=(_Float16)c3.x; a[13]=(_Float16)c3.y; a[14]=(_Float16)c3.z; a[15]=(_Float16)c3.w;
    return a;
}

// f16 A fragment: two aligned 16B non-temporal vector loads (streamed, read-once)
__device__ __forceinline__ v16h load_a_frag_h(const _Float16* base, int ld, int lane) {
    int row = lane & 15, kb = (lane >> 4) * 8;
    const _Float16* p = base + (size_t)row * ld + kb;
    u32x4 lo = __builtin_nontemporal_load((const u32x4*)p);
    u32x4 hi = __builtin_nontemporal_load((const u32x4*)(p + 16));
    v16h a;
    __builtin_memcpy(&a, &lo, 16);
    __builtin_memcpy((char*)&a + 16, &hi, 16);
    return a;
}

// B fragment from LDS: 16 contiguous halfs per lane (2x ds_load_b128)
__device__ __forceinline__ v16h load_b_frag_lds(const _Float16* tile, int lane) {
    return *(const v16h*)(tile + lane * 16);
}

// LayerNorm over a 16-row tile, producing the f16 A fragment.
// Lane pair (l, l^16) holds the two disjoint halves of row l&15.
__device__ __forceinline__ v16h ln_a_frag(const float* nodes, size_t rowbase, int lane,
                                          const float* g, const float* bb) {
    int r = lane & 15, kb = (lane >> 4) * 8;
    const float* p = nodes + (rowbase + r) * 32 + kb;
    float xv[16];
    float4 c0 = *(const float4*)(p + 0);
    float4 c1 = *(const float4*)(p + 4);
    float4 c2 = *(const float4*)(p + 16);
    float4 c3 = *(const float4*)(p + 20);
    xv[0]=c0.x; xv[1]=c0.y; xv[2]=c0.z; xv[3]=c0.w;
    xv[4]=c1.x; xv[5]=c1.y; xv[6]=c1.z; xv[7]=c1.w;
    xv[8]=c2.x; xv[9]=c2.y; xv[10]=c2.z; xv[11]=c2.w;
    xv[12]=c3.x; xv[13]=c3.y; xv[14]=c3.z; xv[15]=c3.w;
    float s = 0.f, s2 = 0.f;
#pragma unroll
    for (int e = 0; e < 16; e++) { s += xv[e]; s2 += xv[e] * xv[e]; }
    s  += __shfl_xor(s, 16, 32);
    s2 += __shfl_xor(s2, 16, 32);
    float mean = s * (1.f / 32.f);
    float var  = s2 * (1.f / 32.f) - mean * mean;
    float inv  = rsqrtf(var + 1e-5f);
    v16h a;
#pragma unroll
    for (int e = 0; e < 16; e++) {
        int k = kb + (e < 8 ? e : e + 8);
        a[e] = (_Float16)((xv[e] - mean) * inv * g[k] + bb[k]);
    }
    return a;
}

// ---------------------------------------------------------------- small kernels

__global__ void rope_init_k(float* rc, float* rs) {
    int tid = blockIdx.x * blockDim.x + threadIdx.x;
    if (tid >= 17 * 32) return;
    int n = tid >> 5, d = tid & 31;
    float invf = powf(10000.f, -(float)(2 * (d >> 1)) / 32.f);
    float ang = (float)n * invf;
    rc[tid] = cosf(ang);
    rs[tid] = sinf(ang);
}

// pack W(K,N) row-major f32 -> per-tile per-lane contiguous f16 B fragments
__global__ void pack_b_k(const float* __restrict__ W, _Float16* __restrict__ out, int K, int N) {
    int tid = blockIdx.x * blockDim.x + threadIdx.x;
    int total = (K / 32) * (N / 16) * 512;
    if (tid >= total) return;
    int tile = tid >> 9, within = tid & 511;
    int lane = within >> 4, e = within & 15;
    int ntiles = N / 16;
    int kt = tile / ntiles, nt = tile % ntiles;
    int k = kt * 32 + (lane >> 4) * 8 + (e < 8 ? e : e + 8);
    int n = nt * 16 + (lane & 15);
    out[tid] = (_Float16)W[(size_t)k * N + n];
}

// nodes[bt,n,:] = x[b,:,t,n] @ emb_w + emb_b
__global__ void embed_k(const float* __restrict__ x, const float* __restrict__ ew,
                        const float* __restrict__ eb, float* __restrict__ nodes) {
    size_t tid = (size_t)blockIdx.x * blockDim.x + threadIdx.x;
    if (tid >= (size_t)NROWS * 32) return;
    int hid = (int)(tid & 31);
    size_t row = tid >> 5;
    int n = (int)(row % 17);
    size_t bt = row / 17;
    int b = (int)(bt >> 10), t = (int)(bt & 1023);
    float acc = eb[hid];
#pragma unroll
    for (int c = 0; c < 3; c++)
        acc += x[(((size_t)(b * 3 + c)) * 1024 + t) * 17 + n] * ew[c * 32 + hid];
    nodes[row * 32 + hid] = acc;
}

// ---------------------------------------------------------------- layer kernels

// LN1 + fused [wq|wkv] projection (N=768, 48 tiles) + bias + rotary on q,k. Output f16.
__global__ void ln_qkv_rope_k(const float* __restrict__ nodes, const float* __restrict__ g,
                              const float* __restrict__ bln, const _Float16* __restrict__ packB,
                              const float* __restrict__ bq, const float* __restrict__ bkv,
                              const float* __restrict__ rc, const float* __restrict__ rs,
                              _Float16* __restrict__ qkv) {
    __shared__ __align__(64) _Float16 sB[48 * 512];   // 48KB packed B panel
    int tid = threadIdx.x;
    stage_to_lds(sB, packB, 48 * 1024, tid);

    int lane = tid & 31, wave = tid >> 5;
    size_t mtile = (size_t)blockIdx.x * WPB + wave;
    size_t rowbase = mtile * 16;

    v16h a = ln_a_frag(nodes, rowbase, lane, g, bln);

    int nodeIdx[8];
#pragma unroll
    for (int v = 0; v < 8; v++)
        nodeIdx[v] = (int)((rowbase + v + 8 * (lane >> 4)) % 17);

    // rope factors: dd is (lane&15) for even tiles, (lane&15)+16 for odd tiles;
    // rotation sign depends only on lane&1.
    int ddE = lane & 15;
    int sgnOdd = lane & 1;
    float rcE[8], rsE[8], rcO[8], rsO[8];
#pragma unroll
    for (int v = 0; v < 8; v++) {
        int nb = nodeIdx[v] * 32;
        rcE[v] = rc[nb + ddE];      rsE[v] = rs[nb + ddE];
        rcO[v] = rc[nb + ddE + 16]; rsO[v] = rs[nb + ddE + 16];
    }

    for (int nt = 0; nt < 48; nt++) {
        v16h b = load_b_frag_lds(sB + nt * 512, lane);
        v8f c = zero8();
        c = wmma16(a, b, c);
        int n = nt * 16 + (lane & 15);
        float bias = (nt < 16) ? bq[n] : bkv[n - 256];
        float xs[8];
#pragma unroll
        for (int v = 0; v < 8; v++) xs[v] = c[v] + bias;
        if (nt < 32) {  // q or k tile (uniform per wave): interleaved-pair rotary
            bool oddT = (nt & 1) != 0;
#pragma unroll
            for (int v = 0; v < 8; v++) {
                float other = __shfl_xor(xs[v], 1, 32);
                float cz = oddT ? rcO[v] : rcE[v];
                float sz = oddT ? rsO[v] : rsE[v];
                xs[v] = sgnOdd ? (xs[v] * cz + other * sz) : (xs[v] * cz - other * sz);
            }
        }
#pragma unroll
        for (int v = 0; v < 8; v++) {
            size_t row = rowbase + v + 8 * (lane >> 4);
            __builtin_nontemporal_store((_Float16)xs[v], &qkv[row * 768 + n]);
        }
    }
}

// one block per graph, one wave per head; edge tensor reduced to (be, we) scalars
__global__ void attn_k(const _Float16* __restrict__ qkv, const float* __restrict__ be,
                       const float* __restrict__ we, _Float16* __restrict__ attnout) {
    __shared__ float sQ[NHEADS][17][33], sK[NHEADS][17][33], sV[NHEADS][17][33];
    __shared__ float sBE[NHEADS][32], sWE[NHEADS][32];
    int h = threadIdx.x >> 5, lane = threadIdx.x & 31;
    size_t gidx = blockIdx.x;
    const _Float16* base = qkv + gidx * 17 * 768 + h * 32;
    for (int i = 0; i < 17; i++) {
        sQ[h][i][lane] = (float)__builtin_nontemporal_load(&base[(size_t)i * 768 + lane]);
        sK[h][i][lane] = (float)__builtin_nontemporal_load(&base[(size_t)i * 768 + 256 + lane]);
        sV[h][i][lane] = (float)__builtin_nontemporal_load(&base[(size_t)i * 768 + 512 + lane]);
    }
    sBE[h][lane] = be[h * 32 + lane];
    sWE[h][lane] = we[h * 32 + lane];
    __syncthreads();

    const float scale = 0.1767766952966369f;  // 1/sqrt(32)
    int j = lane;
    for (int i = 0; i < 17; i++) {
        unsigned Am = ADJ_MASK[i];
        float acc = 0.f, s0 = 0.f, s1 = 0.f;
        for (int d = 0; d < 32; d++) {
            float qv = sQ[h][i][d];
            s0 += qv * sBE[h][d];
            s1 += qv * sWE[h][d];
            if (j < 17) acc += qv * sK[h][j][d];
        }
        float sim = (j < 17) ? (acc + s0 + (((Am >> j) & 1u) ? s1 : 0.f)) * scale : -1e30f;
        float m = sim;
        for (int off = 16; off; off >>= 1) m = fmaxf(m, __shfl_xor(m, off, 32));
        float ev = (j < 17) ? __expf(sim - m) : 0.f;
        float ssum = ev;
        for (int off = 16; off; off >>= 1) ssum += __shfl_xor(ssum, off, 32);
        float attn = ev / ssum;
        float aw = ((j < 17) && ((Am >> j) & 1u)) ? attn : 0.f;
        for (int off = 16; off; off >>= 1) aw += __shfl_xor(aw, off, 32);
        // output: lane == d now
        int d = lane;
        float o = 0.f;
        for (int j2 = 0; j2 < 17; j2++) o += __shfl(attn, j2, 32) * sV[h][j2][d];
        o += sBE[h][d] + aw * sWE[h][d];
        __builtin_nontemporal_store((_Float16)o, &attnout[(gidx * 17 + i) * 256 + h * 32 + d]);
    }
}

// GEMM (f16 A, K = 32*nKt, N = 32) + bias + gated residual:
// gate = sigmoid([x,res,x-res] @ wg); out = x*gate + res*(1-gate)
__global__ void gemm_gate_k(const _Float16* __restrict__ A, int K, int nKt,
                            const _Float16* __restrict__ packB, const float* __restrict__ bias,
                            const float* __restrict__ wg, const float* __restrict__ res,
                            float* __restrict__ out) {
    __shared__ __align__(64) _Float16 sB[16 * 512];   // up to 16KB (nKt*2 tiles)
    int tid = threadIdx.x;
    stage_to_lds(sB, packB, nKt * 2048, tid);

    int lane = tid & 31, wave = tid >> 5;
    size_t mtile = (size_t)blockIdx.x * WPB + wave;
    size_t rowbase = mtile * 16;
    v8f c0 = zero8(), c1 = zero8();
    for (int kt = 0; kt < nKt; kt++) {
        v16h a = load_a_frag_h(A + rowbase * K + kt * 32, K, lane);
        v16h b0 = load_b_frag_lds(sB + (size_t)(kt * 2 + 0) * 512, lane);
        v16h b1 = load_b_frag_lds(sB + (size_t)(kt * 2 + 1) * 512, lane);
        c0 = wmma16(a, b0, c0);
        c1 = wmma16(a, b1, c1);
    }
    int n0 = lane & 15, n1 = n0 + 16;
    float w0x = wg[n0] + wg[64 + n0], w0r = wg[32 + n0] - wg[64 + n0];
    float w1x = wg[n1] + wg[64 + n1], w1r = wg[32 + n1] - wg[64 + n1];
    float bb0 = bias[n0], bb1 = bias[n1];
#pragma unroll
    for (int v = 0; v < 8; v++) {
        size_t row = rowbase + v + 8 * (lane >> 4);
        float x0 = c0[v] + bb0, x1 = c1[v] + bb1;
        float r0 = res[row * 32 + n0], r1 = res[row * 32 + n1];
        float gp = x0 * w0x + r0 * w0r + x1 * w1x + r1 * w1r;
        gp += __shfl_xor(gp, 1, 32);
        gp += __shfl_xor(gp, 2, 32);
        gp += __shfl_xor(gp, 4, 32);
        gp += __shfl_xor(gp, 8, 32);
        float gt = 1.f / (1.f + __expf(-gp));
        out[row * 32 + n0] = x0 * gt + r0 * (1.f - gt);
        out[row * 32 + n1] = x1 * gt + r1 * (1.f - gt);
    }
}

// LN2 + FF1 (32->128) + exact GELU, output f16
__global__ void ln_ff1_k(const float* __restrict__ nodes, const float* __restrict__ g,
                         const float* __restrict__ bln, const _Float16* __restrict__ packB,
                         const float* __restrict__ b1, _Float16* __restrict__ ff) {
    __shared__ __align__(64) _Float16 sB[8 * 512];    // 8KB
    int tid = threadIdx.x;
    stage_to_lds(sB, packB, 8 * 1024, tid);

    int lane = tid & 31, wave = tid >> 5;
    size_t mtile = (size_t)blockIdx.x * WPB + wave;
    size_t rowbase = mtile * 16;
    v16h a = ln_a_frag(nodes, rowbase, lane, g, bln);
    for (int nt = 0; nt < 8; nt++) {
        v16h b = load_b_frag_lds(sB + (size_t)nt * 512, lane);
        v8f c = zero8();
        c = wmma16(a, b, c);
        int n = nt * 16 + (lane & 15);
        float bias = b1[n];
#pragma unroll
        for (int v = 0; v < 8; v++) {
            size_t row = rowbase + v + 8 * (lane >> 4);
            float x = c[v] + bias;
            x = 0.5f * x * (1.f + erff(x * 0.7071067811865476f));
            __builtin_nontemporal_store((_Float16)x, &ff[row * 128 + n]);
        }
    }
}

// ---------------------------------------------------------------- pooling + MLP

__global__ void reduce_k(const float* __restrict__ nodes, float* __restrict__ feat) {
    __shared__ float red[256];
    int b = blockIdx.x, tid = threadIdx.x;
    int c = tid & 31, part = tid >> 5;  // 8 partials per channel
    float s = 0.f;
    for (int r = part; r < 17408; r += 8)
        s += nodes[((size_t)b * 17408 + r) * 32 + c];
    red[tid] = s;
    __syncthreads();
    if (part == 0) {
        float t = 0.f;
        for (int p = 0; p < 8; p++) t += red[p * 32 + c];
        feat[b * 32 + c] = t / 17408.f;
    }
}

__global__ void mlp_k(const float* __restrict__ feat, const float* __restrict__ w1,
                      const float* __restrict__ b1, const float* __restrict__ w2,
                      const float* __restrict__ b2, float* __restrict__ out) {
    __shared__ float f[32];
    __shared__ float mid[64];
    int b = blockIdx.x, t = threadIdx.x;
    if (t < 32) f[t] = feat[b * 32 + t];
    __syncthreads();
    float m = b1[t];
    for (int c = 0; c < 32; c++) m += f[c] * w1[c * 64 + t];
    mid[t] = fmaxf(m, 0.f);
    __syncthreads();
    if (t < 32) {
        float o = b2[t];
        for (int j = 0; j < 64; j++) o += mid[j] * w2[j * 32 + t];
        out[b * 32 + t] = o;
    }
}

// ---------------------------------------------------------------- host launcher

extern "C" void kernel_launch(void* const* d_in, const int* in_sizes, int n_in,
                              void* d_out, int out_size, void* d_ws, size_t ws_size,
                              hipStream_t stream) {
    const float* X      = (const float*)d_in[0];
    const float* emb_w  = (const float*)d_in[1];
    const float* emb_b  = (const float*)d_in[2];
    const float* mlp_w1 = (const float*)d_in[3];
    const float* mlp_b1 = (const float*)d_in[4];
    const float* mlp_w2 = (const float*)d_in[5];
    const float* mlp_b2 = (const float*)d_in[6];
    auto LP = [&](int l, int k) { return (const float*)d_in[7 + 18 * l + k]; };
    // per-layer param order: 0 ln1_g 1 ln1_b 2 wq 3 bq 4 wkv 5 bkv 6 we 7 be
    //                        8 wo 9 bo 10 wg1 11 ln2_g 12 ln2_b 13 w1 14 b1 15 w2 16 b2 17 wg2

    char* ws = (char*)d_ws;
    float*     nodes_a = (float*)(ws + 0ull);                     // 35,651,584 B
    float*     nodes_b = (float*)(ws + 35651584ull);              // 35,651,584 B
    _Float16*  qkv     = (_Float16*)(ws + 71303168ull);           // 427,819,008 B (f16)
    _Float16*  ffb     = (_Float16*)(ws + 71303168ull);           // alias: used after qkv consumed
    _Float16*  attn    = (_Float16*)(ws + 499122176ull);          // 142,606,336 B (f16)
    char*      packb   = ws + 641728512ull;                       // 4 * 81,920 B
    float*     ropeC   = (float*)(ws + 642056192ull);             // 544 f32
    float*     ropeS   = ropeC + 544;
    float*     feat    = (float*)(ws + 642060544ull);             // 16x32 f32

    rope_init_k<<<3, 256, 0, stream>>>(ropeC, ropeS);
    embed_k<<<(NROWS * 32) / 256, 256, 0, stream>>>(X, emb_w, emb_b, nodes_a);

    for (int l = 0; l < 4; l++) {
        _Float16* pk      = (_Float16*)(packb + (size_t)l * 81920);
        _Float16* pk_qkv  = pk;             // 48 tiles (wq 16 + wkv 32)
        _Float16* pk_wo   = pk + 48 * 512;  // 16 tiles (8 ktiles x 2 ntiles)
        _Float16* pk_w1   = pk + 64 * 512;  // 8 tiles
        _Float16* pk_w2   = pk + 72 * 512;  // 8 tiles (4 ktiles x 2 ntiles)

        pack_b_k<<<32, 256, 0, stream>>>(LP(l, 2),  pk_qkv,            32, 256);  // wq
        pack_b_k<<<64, 256, 0, stream>>>(LP(l, 4),  pk_qkv + 16 * 512, 32, 512);  // wkv
        pack_b_k<<<32, 256, 0, stream>>>(LP(l, 8),  pk_wo,            256, 32);   // wo
        pack_b_k<<<16, 256, 0, stream>>>(LP(l, 13), pk_w1,             32, 128);  // w1
        pack_b_k<<<16, 256, 0, stream>>>(LP(l, 15), pk_w2,            128, 32);   // w2

        ln_qkv_rope_k<<<MT_BLOCKS, 256, 0, stream>>>(nodes_a, LP(l, 0), LP(l, 1), pk_qkv,
                                                     LP(l, 3), LP(l, 5), ropeC, ropeS, qkv);
        attn_k<<<NBT, 256, 0, stream>>>(qkv, LP(l, 7), LP(l, 6), attn);
        gemm_gate_k<<<MT_BLOCKS, 256, 0, stream>>>(attn, 256, 8, pk_wo, LP(l, 9), LP(l, 10),
                                                   nodes_a, nodes_b);
        ln_ff1_k<<<MT_BLOCKS, 256, 0, stream>>>(nodes_b, LP(l, 11), LP(l, 12), pk_w1,
                                                LP(l, 14), ffb);
        gemm_gate_k<<<MT_BLOCKS, 256, 0, stream>>>(ffb, 128, 4, pk_w2, LP(l, 16), LP(l, 17),
                                                   nodes_b, nodes_a);
    }

    reduce_k<<<16, 256, 0, stream>>>(nodes_a, feat);
    mlp_k<<<16, 64, 0, stream>>>(feat, mlp_w1, mlp_b1, mlp_w2, mlp_b2, (float*)d_out);
}
